// LSTMNet_67250597921538
// MI455X (gfx1250) — compile-verified, matrix operand order checked by reading, not run on hardware
//
#include <hip/hip_runtime.h>

// LSTM: B=512, T=1024, H=100, input_size=1.
// 32 workgroups x 16 batch rows; each WG runs the whole recurrence.
// Per step: gates(16x512) = [h | x_t | 1 | 0](16x128) @ W_aug(128x512) via
// v_wmma_f32_16x16x32_f16; W_aug lives in VGPRs (B-fragments) for all steps.
// h recirculates through LDS; c stays in registers; x slice staged in LDS.
// Activations use V_TANH_F32 (gfx1250 TRANS op) -- no precise divides.

#define H     100
#define TLEN  1024
#define ROWS  16          // batch rows per workgroup
#define GSTR  516         // gate LDS row stride (f32), padded vs 512
#define HSTR  136         // h LDS row stride (f16), padded vs 128
#define XSTR  1032        // x LDS row stride (f16), padded vs 1024 (+1 slack col)

typedef __attribute__((ext_vector_type(16))) _Float16 v16h;
typedef __attribute__((ext_vector_type(4)))  _Float16 v4h;
typedef __attribute__((ext_vector_type(8)))  float    v8f;

__device__ __forceinline__ float fast_tanh(float x) {
#if __has_builtin(__builtin_amdgcn_tanhf)
    return __builtin_amdgcn_tanhf(x);            // v_tanh_f32: 1 TRANS op
#else
    const float xc = fminf(fmaxf(x, -15.f), 15.f);
    const float e  = __builtin_amdgcn_exp2f(xc * 2.8853900817779268f); // exp(2x)
    return 1.f - 2.f * __builtin_amdgcn_rcpf(e + 1.f);                 // no v_div
#endif
}
__device__ __forceinline__ float fast_sigm(float x) {
    return fmaf(0.5f, fast_tanh(0.5f * x), 0.5f);
}

// Split workgroup barrier: signal (with LDS drain), then wait. Work placed
// between the two is hidden in the barrier slack.
#define WG_SIGNAL() asm volatile("s_wait_dscnt 0x0\n\ts_barrier_signal -1" ::: "memory")
#define WG_WAIT()   asm volatile("s_barrier_wait -1" ::: "memory")

__global__ __launch_bounds__(512)
void lstm_wmma_kernel(const float* __restrict__ xin,    // [512,1024]
                      const float* __restrict__ W_ih,   // [400]
                      const float* __restrict__ W_hh,   // [400,100]
                      const float* __restrict__ b_ih,   // [400]
                      const float* __restrict__ b_hh,   // [400]
                      const float* __restrict__ W_out,  // [100]
                      const float* __restrict__ b_out,  // [1]
                      float* __restrict__ out)          // [512,1024]
{
    __shared__ __align__(16) _Float16 hbuf[ROWS][HSTR]; // augmented h tile (f16)
    __shared__ __align__(16) float    gbuf[ROWS][GSTR]; // gate accumulators (f32)
    __shared__ __align__(16) _Float16 xlds[ROWS][XSTR]; // staged input slice (f16)

    const int tid     = threadIdx.x;
    const int wave    = tid >> 5;        // 0..15 : owns N-tiles 2w,2w+1 AND batch row w
    const int lane    = tid & 31;
    const int half    = lane >> 4;
    const int c16     = lane & 15;
    const int rowBase = blockIdx.x * ROWS;

    // ---- stage this WG's input slice into LDS (f16), coalesced ------------
    for (int i = tid; i < ROWS * XSTR; i += 512) {
        const int r = i / XSTR, t = i - r * XSTR;
        xlds[r][t] = (_Float16)((t < TLEN) ? xin[(rowBase + r) * TLEN + t] : 0.f);
    }

    // ---- stationary B fragments (augmented weights, f16) ------------------
    // B[k][n]: n = gate*128 + j (j<100), k<100 -> W_hh, k==100 -> W_ih,
    // k==101 -> b_ih+b_hh, else 0.
    auto bval = [&](int k, int n) -> float {
        const int g = n >> 7, j = n & 127;
        if (j >= H) return 0.f;
        const int r = g * H + j;
        if (k <  H)     return W_hh[r * H + k];
        if (k == H)     return W_ih[r];
        if (k == H + 1) return b_ih[r] + b_hh[r];
        return 0.f;
    };
    v16h bfrag[2][4];   // [n-tile][k-chunk]; 64 VGPRs/lane, resident all steps
#pragma unroll
    for (int nt = 0; nt < 2; ++nt) {
        const int n = (2 * wave + nt) * 16 + c16;
#pragma unroll
        for (int kc = 0; kc < 4; ++kc) {
            v16h f;
#pragma unroll
            for (int v = 0; v < 8; ++v) {
                const int k = kc * 32 + ((v >= 4) ? 16 : 0) + half * 8 + (v & 3) * 2;
                f[2 * v]     = (_Float16)bval(k, n);
                f[2 * v + 1] = (_Float16)bval(k + 1, n);
            }
            bfrag[nt][kc] = f;
        }
    }

    // ---- per-lane cell slots: row == wave, cols = lane*4 .. lane*4+3 ------
    const int col0 = lane << 2;
    float m_h[4], m_x[4], m_one[4], wout[4], creg[4];
#pragma unroll
    for (int s = 0; s < 4; ++s) {
        const int c = col0 + s;
        m_h[s]   = (c <  H)     ? 1.f : 0.f;
        m_x[s]   = (c == H)     ? 1.f : 0.f;
        m_one[s] = (c == H + 1) ? 1.f : 0.f;
        wout[s]  = (c <  H)     ? W_out[c] : 0.f;
        creg[s]  = 0.f;
    }
    const float bout = b_out[0];

    // init augmented h tile: h0 = 0, col100 = x[.,0], col101 = 1, pads 0
    {
        const float x0 = xin[(rowBase + wave) * TLEN];
        v4h hv;
#pragma unroll
        for (int s = 0; s < 4; ++s)
            hv[s] = (_Float16)(m_x[s] * x0 + m_one[s]);
        *reinterpret_cast<v4h*>(&hbuf[wave][col0]) = hv;
    }
    __syncthreads();

    // ---------------- recurrence ------------------------------------------
    const _Float16* arow = &hbuf[c16][half * 8];
    for (int t = 0; t < TLEN; ++t) {
        // A fragments: four independent buffers, all ds_load_b128 before WMMA
        v16h a0, a1, a2, a3;
        reinterpret_cast<int4*>(&a0)[0] = *reinterpret_cast<const int4*>(arow +  0);
        reinterpret_cast<int4*>(&a0)[1] = *reinterpret_cast<const int4*>(arow + 16);
        reinterpret_cast<int4*>(&a1)[0] = *reinterpret_cast<const int4*>(arow + 32);
        reinterpret_cast<int4*>(&a1)[1] = *reinterpret_cast<const int4*>(arow + 48);
        reinterpret_cast<int4*>(&a2)[0] = *reinterpret_cast<const int4*>(arow + 64);
        reinterpret_cast<int4*>(&a2)[1] = *reinterpret_cast<const int4*>(arow + 80);
        reinterpret_cast<int4*>(&a3)[0] = *reinterpret_cast<const int4*>(arow + 96);
        reinterpret_cast<int4*>(&a3)[1] = *reinterpret_cast<const int4*>(arow + 112);

        v8f acc[2] = {};
#pragma unroll
        for (int nt = 0; nt < 2; ++nt) {
            acc[nt] = __builtin_amdgcn_wmma_f32_16x16x32_f16(
                false, a0, false, bfrag[nt][0], (short)0, acc[nt], false, false);
            acc[nt] = __builtin_amdgcn_wmma_f32_16x16x32_f16(
                false, a1, false, bfrag[nt][1], (short)0, acc[nt], false, false);
            acc[nt] = __builtin_amdgcn_wmma_f32_16x16x32_f16(
                false, a2, false, bfrag[nt][2], (short)0, acc[nt], false, false);
            acc[nt] = __builtin_amdgcn_wmma_f32_16x16x32_f16(
                false, a3, false, bfrag[nt][3], (short)0, acc[nt], false, false);
        }

        // scatter gates (D layout: VGPR j -> M = j + 8*half, N = lane%16)
#pragma unroll
        for (int nt = 0; nt < 2; ++nt) {
            const int n = (2 * wave + nt) * 16 + c16;
#pragma unroll
            for (int j = 0; j < 8; ++j)
                gbuf[j + 8 * half][n] = acc[nt][j];
        }
        WG_SIGNAL();            // gates drained to LDS
        WG_WAIT();

        // branch-free cell update: gate cols >=100 are exactly 0 (zero B cols)
        float gi4[4], gf4[4], gg4[4], go4[4];
        *reinterpret_cast<float4*>(gi4) = *reinterpret_cast<const float4*>(&gbuf[wave][      col0]);
        *reinterpret_cast<float4*>(gf4) = *reinterpret_cast<const float4*>(&gbuf[wave][128 + col0]);
        *reinterpret_cast<float4*>(gg4) = *reinterpret_cast<const float4*>(&gbuf[wave][256 + col0]);
        *reinterpret_cast<float4*>(go4) = *reinterpret_cast<const float4*>(&gbuf[wave][384 + col0]);
        const float xv = (float)xlds[wave][t + 1];   // x for step t+1 (0 past end)

        float p = 0.f;
        v4h hv;
#pragma unroll
        for (int s = 0; s < 4; ++s) {
            const float gi = fast_sigm(gi4[s]);
            const float gf = fast_sigm(gf4[s]);
            const float gg = fast_tanh(gg4[s]);
            const float go = fast_sigm(go4[s]);
            const float cn = gf * creg[s] + gi * gg;
            creg[s] = cn;
            const float hn = go * fast_tanh(cn);
            hv[s] = (_Float16)fmaf(m_h[s], hn, fmaf(m_x[s], xv, m_one[s]));
            p = fmaf(hn, wout[s], p);
        }
        *reinterpret_cast<v4h*>(&hbuf[wave][col0]) = hv;   // one ds_store_b64

        WG_SIGNAL();            // h drained; hide output reduction in barrier
        p += __shfl_xor(p, 16, 32);
        p += __shfl_xor(p,  8, 32);
        p += __shfl_xor(p,  4, 32);
        p += __shfl_xor(p,  2, 32);
        p += __shfl_xor(p,  1, 32);
        if (lane == 0)
            out[(rowBase + wave) * TLEN + t] = p + bout;
        WG_WAIT();
    }
}

extern "C" void kernel_launch(void* const* d_in, const int* in_sizes, int n_in,
                              void* d_out, int out_size, void* d_ws, size_t ws_size,
                              hipStream_t stream) {
    const float* xin   = (const float*)d_in[0];
    const float* W_ih  = (const float*)d_in[1];
    const float* W_hh  = (const float*)d_in[2];
    const float* b_ih  = (const float*)d_in[3];
    const float* b_hh  = (const float*)d_in[4];
    const float* W_out = (const float*)d_in[5];
    const float* b_out = (const float*)d_in[6];
    float* outp = (float*)d_out;

    lstm_wmma_kernel<<<dim3(512 / ROWS), dim3(512), 0, stream>>>(
        xin, W_ih, W_hh, b_ih, b_hh, W_out, b_out, outp);
}
